// ALiBiMultiHeadAttention_28252294873711
// MI455X (gfx1250) — matrix operand order repaired and checked by reading, not compiled, not run on hardware
//
#include <hip/hip_runtime.h>
#include <hip/hip_bf16.h>

// ---------------------------------------------------------------------------
// ALiBi Multi-Head Attention for MI455X (gfx1250, wave32, WMMA)
// B=2, T=2048, D=1024, H=16, DK=64
//  * alibi_bias tensor (256MB) never read: bias = slope_h*(j-i) in-register.
//  * f16 intermediates + v_wmma_f32_16x16x32_f16 (K=32 per wmma).
//  * Flash-attention online softmax, per-wave 16 query rows.
//  * Round-2 fixes: 32 rows/wave in GEMMs (each B-frag feeds 2 WMMAs),
//    batched B-frag LDS loads before WMMA chains, branchless prefetch.
// ---------------------------------------------------------------------------

typedef __attribute__((ext_vector_type(16))) _Float16 v16h;
typedef __attribute__((ext_vector_type(8)))  float    v8f;

static constexpr int Bc = 2;
static constexpr int Tc = 2048;
static constexpr int Dc = 1024;
static constexpr int Hc = 16;
static constexpr int DKc = 64;

struct __align__(16) H8 { _Float16 h[8]; };

__device__ __forceinline__ v8f wmma_f16(v16h a, v16h b, v8f c) {
  // 8 args: (neg_a, A, neg_b, B, c_mod, C, reuse_a, reuse_b)
  return __builtin_amdgcn_wmma_f32_16x16x32_f16(false, a, false, b,
                                                (short)0, c, false, false);
}

// ---------------------------------------------------------------------------
// QKV projection GEMM: Y[m,n] = sum_k X[m,k] * W[n,k]  (X @ W^T)
// Block: 128 threads (4 waves), 128x64 tile; each wave owns 32 rows
// (2 A-frags), so each of the 4 B-frags feeds 2 WMMAs (8 WMMA / wave / slab).
// Output f16 in [B,H,T,DK]; n0 multiple of DK -> h/b/t block-uniform.
// ---------------------------------------------------------------------------

struct QKVArgs {
  const float* x[3];
  const float* w[3];
  _Float16*    o[3];
};

__global__ __launch_bounds__(128) void proj_qkv_kernel(QKVArgs args) {
  constexpr int AS = 48;                        // LDS row stride (halves), 96B
  __shared__ __align__(16) _Float16 sA[128 * AS];
  __shared__ __align__(16) _Float16 sB[64 * AS];

  const float* __restrict__ X = args.x[blockIdx.z];
  const float* __restrict__ W = args.w[blockIdx.z];
  _Float16* __restrict__    O = args.o[blockIdx.z];

  const int tid  = threadIdx.x;
  const int n0   = blockIdx.x * 64;
  const int m0   = blockIdx.y * 128;
  const int wave = tid >> 5;
  const int lane = tid & 31;
  const int hlf  = lane >> 4;
  const int l16  = lane & 15;
  const int klo  = hlf ? 16 : 0;                // contiguous k = 16*hlf + e
  const int khi  = hlf ? 24 : 8;

  const int srow = tid >> 3;                    // staging: 8 chunks/row of 32
  const int scol = (tid & 7) * 4;

  v8f acc[2][4];
#pragma unroll
  for (int s = 0; s < 2; s++)
#pragma unroll
    for (int i = 0; i < 4; i++)
#pragma unroll
      for (int e = 0; e < 8; e++) acc[s][i][e] = 0.0f;

  for (int k0 = 0; k0 < Dc; k0 += 32) {
    // ---- Stage: 8 A-chunks (128x32) + 4 B-chunks (64x32), loads first.
    float4 ta[8], tb[4];
#pragma unroll
    for (int i = 0; i < 8; i++)
      ta[i] = *(const float4*)(X + (size_t)(m0 + srow + i * 16) * Dc + k0 + scol);
#pragma unroll
    for (int i = 0; i < 4; i++)
      tb[i] = *(const float4*)(W + (size_t)(n0 + srow + i * 16) * Dc + k0 + scol);
    {
      const int kp = (k0 + 32 < Dc) ? (k0 + 32) : k0;   // branchless prefetch
      __builtin_prefetch(X + (size_t)(m0 + srow) * Dc + kp + scol, 0, 1);
      __builtin_prefetch(W + (size_t)(n0 + srow) * Dc + kp + scol, 0, 1);
    }
#pragma unroll
    for (int i = 0; i < 8; i++) {
      _Float16* pa = &sA[(srow + i * 16) * AS + scol];
      pa[0] = (_Float16)ta[i].x; pa[1] = (_Float16)ta[i].y;
      pa[2] = (_Float16)ta[i].z; pa[3] = (_Float16)ta[i].w;
    }
#pragma unroll
    for (int i = 0; i < 4; i++) {
      _Float16* pb = &sB[(srow + i * 16) * AS + scol];
      pb[0] = (_Float16)tb[i].x; pb[1] = (_Float16)tb[i].y;
      pb[2] = (_Float16)tb[i].z; pb[3] = (_Float16)tb[i].w;
    }
    __syncthreads();

    // ---- Compute: 2 A-frags + 4 batched B-frags -> 8 WMMAs per wave.
    v16h af[2];
#pragma unroll
    for (int s = 0; s < 2; s++) {
      const int mrow = wave * 32 + s * 16 + l16;
#pragma unroll
      for (int e = 0; e < 8; e++) {
        af[s][e]     = sA[mrow * AS + klo + e];
        af[s][8 + e] = sA[mrow * AS + khi + e];
      }
    }
    v16h bf[4];
#pragma unroll
    for (int nt = 0; nt < 4; nt++) {
      const int n  = nt * 16 + l16;
      const int kb = hlf * 16;
#pragma unroll
      for (int e = 0; e < 16; e++) bf[nt][e] = sB[n * AS + kb + e];
    }
#pragma unroll
    for (int nt = 0; nt < 4; nt++) {
      acc[0][nt] = wmma_f16(af[0], bf[nt], acc[0][nt]);
      acc[1][nt] = wmma_f16(af[1], bf[nt], acc[1][nt]);
    }
    __syncthreads();
  }

  // ---- Store f16 [B,H,T,DK]; block-uniform h = n0/DK, b = m0/T.
  const int h    = n0 / DKc;
  const int bidx = m0 / Tc;
#pragma unroll
  for (int s = 0; s < 2; s++) {
    const int t0 = (m0 % Tc) + wave * 32 + s * 16 + hlf * 8;
    _Float16* __restrict__ obase =
        O + (size_t)((bidx * Hc + h) * Tc + t0) * DKc + l16;
#pragma unroll
    for (int nt = 0; nt < 4; nt++)
#pragma unroll
      for (int r = 0; r < 8; r++)
        obase[r * DKc + nt * 16] = (_Float16)acc[s][nt][r];
  }
}

// ---------------------------------------------------------------------------
// Output projection: out[m,n] = sum_k A[m,k]*Wo[n,k] + bo[n], f32 output.
// Same 128x64 tiling as the QKV GEMM.
// ---------------------------------------------------------------------------

__global__ __launch_bounds__(128) void out_proj_kernel(
    const _Float16* __restrict__ Ain, const float* __restrict__ W,
    const float* __restrict__ bias, float* __restrict__ Out) {
  constexpr int AS = 48;
  __shared__ __align__(16) _Float16 sA[128 * AS];
  __shared__ __align__(16) _Float16 sB[64 * AS];

  const int tid  = threadIdx.x;
  const int n0   = blockIdx.x * 64;
  const int m0   = blockIdx.y * 128;
  const int wave = tid >> 5;
  const int lane = tid & 31;
  const int hlf  = lane >> 4;
  const int l16  = lane & 15;
  const int klo  = hlf ? 16 : 0;
  const int khi  = hlf ? 24 : 8;

  const int srow = tid >> 3;                    // W staging
  const int scol = (tid & 7) * 4;
  const int arow = tid >> 2;                    // A staging (f16, H8 chunks)
  const int acol = (tid & 3) * 8;

  float bfv[4];
#pragma unroll
  for (int nt = 0; nt < 4; nt++) bfv[nt] = bias[n0 + nt * 16 + l16];

  v8f acc[2][4];
#pragma unroll
  for (int s = 0; s < 2; s++)
#pragma unroll
    for (int i = 0; i < 4; i++)
#pragma unroll
      for (int e = 0; e < 8; e++) acc[s][i][e] = 0.0f;

  for (int k0 = 0; k0 < Dc; k0 += 32) {
    H8 tah[4];
    float4 tb[4];
#pragma unroll
    for (int i = 0; i < 4; i++)
      tah[i] = *(const H8*)(Ain + (size_t)(m0 + arow + i * 32) * Dc + k0 + acol);
#pragma unroll
    for (int i = 0; i < 4; i++)
      tb[i] = *(const float4*)(W + (size_t)(n0 + srow + i * 16) * Dc + k0 + scol);
    {
      const int kp = (k0 + 32 < Dc) ? (k0 + 32) : k0;
      __builtin_prefetch(Ain + (size_t)(m0 + arow) * Dc + kp + acol, 0, 1);
      __builtin_prefetch(W + (size_t)(n0 + srow) * Dc + kp + scol, 0, 1);
    }
#pragma unroll
    for (int i = 0; i < 4; i++)
      *(H8*)(&sA[(arow + i * 32) * AS + acol]) = tah[i];
#pragma unroll
    for (int i = 0; i < 4; i++) {
      _Float16* pb = &sB[(srow + i * 16) * AS + scol];
      pb[0] = (_Float16)tb[i].x; pb[1] = (_Float16)tb[i].y;
      pb[2] = (_Float16)tb[i].z; pb[3] = (_Float16)tb[i].w;
    }
    __syncthreads();

    v16h af[2];
#pragma unroll
    for (int s = 0; s < 2; s++) {
      const int mrow = wave * 32 + s * 16 + l16;
#pragma unroll
      for (int e = 0; e < 8; e++) {
        af[s][e]     = sA[mrow * AS + klo + e];
        af[s][8 + e] = sA[mrow * AS + khi + e];
      }
    }
    v16h bf[4];
#pragma unroll
    for (int nt = 0; nt < 4; nt++) {
      const int n  = nt * 16 + l16;
      const int kb = hlf * 16;
#pragma unroll
      for (int e = 0; e < 16; e++) bf[nt][e] = sB[n * AS + kb + e];
    }
#pragma unroll
    for (int nt = 0; nt < 4; nt++) {
      acc[0][nt] = wmma_f16(af[0], bf[nt], acc[0][nt]);
      acc[1][nt] = wmma_f16(af[1], bf[nt], acc[1][nt]);
    }
    __syncthreads();
  }

#pragma unroll
  for (int s = 0; s < 2; s++) {
    float* __restrict__ obase =
        Out + (size_t)(m0 + wave * 32 + s * 16 + hlf * 8) * Dc + n0 + l16;
#pragma unroll
    for (int nt = 0; nt < 4; nt++)
#pragma unroll
      for (int r = 0; r < 8; r++)
        obase[r * Dc + nt * 16] = acc[s][nt][r] + bfv[nt];
  }
}

// ---------------------------------------------------------------------------
// Flash attention with analytic ALiBi bias + causal mask.
// Grid: (T/64, H, B). Block: 128 threads = 4 waves, each wave owns 16 q-rows.
// ---------------------------------------------------------------------------

__global__ __launch_bounds__(128) void attn_kernel(
    const _Float16* __restrict__ Qh, const _Float16* __restrict__ Kh,
    const _Float16* __restrict__ Vh, _Float16* __restrict__ Oh) {
  constexpr int VS = 72;                        // LDS row stride (halves), 144B
  __shared__ __align__(16) _Float16 sK[64 * VS];    // K tile, [j][d]
  __shared__ __align__(16) _Float16 sVt[64 * VS];   // V tile transposed, [d][j]
  __shared__ __align__(16) _Float16 sP[4][16 * VS]; // per-wave P scratch

  const int tid  = threadIdx.x;
  const int wave = tid >> 5;
  const int lane = tid & 31;
  const int hlf  = lane >> 4;
  const int l16  = lane & 15;
  const int klo  = hlf ? 16 : 0;
  const int khi  = hlf ? 24 : 8;

  const int q0 = blockIdx.x * 64;
  const int h  = blockIdx.y;
  const int b  = blockIdx.z;
  const size_t headoff = (size_t)(b * Hc + h) * Tc * DKc;
  const _Float16* __restrict__ Qb = Qh + headoff;
  const _Float16* __restrict__ Kb = Kh + headoff;
  const _Float16* __restrict__ Vb = Vh + headoff;

  const float slope = exp2f(-0.5f * (float)(h + 1));  // 2^(-8*(h+1)/H), H=16

  const int krow = tid >> 3;                    // staging: rows 0..15 (+16*i)
  const int kcol = (tid & 7) * 8;

  // Q fragments: this wave's 16 rows (16x64 -> two 16x32 A-frags) in registers.
  v16h qa[2];
  {
    const _Float16* qrow = Qb + (size_t)(q0 + wave * 16 + l16) * DKc;
#pragma unroll
    for (int f = 0; f < 2; f++) {
      const H8 lo = *(const H8*)(qrow + f * 32 + klo);
      const H8 hi = *(const H8*)(qrow + f * 32 + khi);
#pragma unroll
      for (int e = 0; e < 8; e++) {
        qa[f][e]     = lo.h[e];
        qa[f][8 + e] = hi.h[e];
      }
    }
  }

  float mI[8], lI[8];
  v8f oacc[4];
#pragma unroll
  for (int r = 0; r < 8; r++) { mI[r] = -1e30f; lI[r] = 0.0f; }
#pragma unroll
  for (int dt = 0; dt < 4; dt++)
#pragma unroll
    for (int e = 0; e < 8; e++) oacc[dt][e] = 0.0f;

  const int nkt = q0 / 64 + 1;                  // causal: only tiles k0 <= q0
  for (int kt = 0; kt < nkt; kt++) {
    const int k0 = kt * 64;

    // ---- Stage K (row-major) + V (transposed): loads first, then stores.
    H8 tk[4], tv[4];
#pragma unroll
    for (int i = 0; i < 4; i++) {
      tk[i] = *(const H8*)(Kb + (size_t)(k0 + krow + i * 16) * DKc + kcol);
      tv[i] = *(const H8*)(Vb + (size_t)(k0 + krow + i * 16) * DKc + kcol);
    }
    {
      const int kp = (kt + 1 < nkt) ? (k0 + 64) : k0;   // branchless prefetch
      __builtin_prefetch(Kb + (size_t)(kp + krow) * DKc + kcol, 0, 1);
      __builtin_prefetch(Vb + (size_t)(kp + krow) * DKc + kcol, 0, 1);
    }
#pragma unroll
    for (int i = 0; i < 4; i++) {
      *(H8*)(&sK[(krow + i * 16) * VS + kcol]) = tk[i];
#pragma unroll
      for (int e = 0; e < 8; e++)
        sVt[(kcol + e) * VS + krow + i * 16] = tv[i].h[e];
    }
    __syncthreads();

    // ---- S = (Q K^T)/8 + slope*(j-i), causal mask.
    // Pairs of j-tiles: 8 batched ds_loads, then 4 WMMAs.
    v8f p[4];
    float rmax[8];
#pragma unroll
    for (int r = 0; r < 8; r++) rmax[r] = -1e30f;

#pragma unroll
    for (int jp = 0; jp < 2; jp++) {
      v16h bfa[2][2];
#pragma unroll
      for (int u = 0; u < 2; u++) {
        const int j  = (jp * 2 + u) * 16 + l16;
        const int kb = hlf * 16;
#pragma unroll
        for (int e = 0; e < 16; e++) {
          bfa[u][0][e] = sK[j * VS + kb + e];        // d 0..31
          bfa[u][1][e] = sK[j * VS + 32 + kb + e];   // d 32..63
        }
      }
#pragma unroll
      for (int u = 0; u < 2; u++) {
        const int jt = jp * 2 + u;
        v8f s;
#pragma unroll
        for (int e = 0; e < 8; e++) s[e] = 0.0f;
        s = wmma_f16(qa[0], bfa[u][0], s);
        s = wmma_f16(qa[1], bfa[u][1], s);

        const int jj = k0 + jt * 16 + l16;           // key index (lane = col)
#pragma unroll
        for (int r = 0; r < 8; r++) {
          const int i = q0 + wave * 16 + hlf * 8 + r;
          const float v = s[r] * 0.125f + slope * (float)(jj - i);
          s[r] = (jj > i) ? -1e9f : v;
          rmax[r] = fmaxf(rmax[r], s[r]);
        }
        p[jt] = s;
      }
    }

    // Row-max across the 16 lanes of each half-wave (masks 1..8 stay in-half).
#pragma unroll
    for (int r = 0; r < 8; r++)
#pragma unroll
      for (int m = 1; m < 16; m <<= 1)
        rmax[r] = fmaxf(rmax[r], __shfl_xor(rmax[r], m, 32));

    float corr[8];
#pragma unroll
    for (int r = 0; r < 8; r++) {
      const float mnew = fmaxf(mI[r], rmax[r]);
      corr[r] = __expf(mI[r] - mnew);
      mI[r]   = mnew;
    }

    float rsum[8];
#pragma unroll
    for (int r = 0; r < 8; r++) rsum[r] = 0.0f;
#pragma unroll
    for (int jt = 0; jt < 4; jt++)
#pragma unroll
      for (int r = 0; r < 8; r++) {
        const float e = __expf(p[jt][r] - mI[r]);
        p[jt][r] = e;
        rsum[r] += e;
      }
#pragma unroll
    for (int r = 0; r < 8; r++) {
#pragma unroll
      for (int m = 1; m < 16; m <<= 1) rsum[r] += __shfl_xor(rsum[r], m, 32);
      lI[r] = lI[r] * corr[r] + rsum[r];
    }
#pragma unroll
    for (int dt = 0; dt < 4; dt++)
#pragma unroll
      for (int r = 0; r < 8; r++) oacc[dt][r] *= corr[r];

    // ---- P (C-layout) -> per-wave LDS scratch -> A-fragments.
    _Float16* myP = &sP[wave][0];
#pragma unroll
    for (int jt = 0; jt < 4; jt++)
#pragma unroll
      for (int r = 0; r < 8; r++)
        myP[(hlf * 8 + r) * VS + jt * 16 + l16] = (_Float16)p[jt][r];

    v16h pa[2];
#pragma unroll
    for (int f = 0; f < 2; f++)
#pragma unroll
      for (int e = 0; e < 8; e++) {
        pa[f][e]     = myP[l16 * VS + f * 32 + klo + e];
        pa[f][8 + e] = myP[l16 * VS + f * 32 + khi + e];
      }

    // ---- O += P V: pairs of d-tiles, batched ds_loads then 4 WMMAs.
#pragma unroll
    for (int dp = 0; dp < 2; dp++) {
      v16h bfa[2][2];
#pragma unroll
      for (int u = 0; u < 2; u++) {
        const int d  = (dp * 2 + u) * 16 + l16;
        const int kb = hlf * 16;
#pragma unroll
        for (int e = 0; e < 16; e++) {
          bfa[u][0][e] = sVt[d * VS + kb + e];       // j 0..31
          bfa[u][1][e] = sVt[d * VS + 32 + kb + e];  // j 32..63
        }
      }
#pragma unroll
      for (int u = 0; u < 2; u++) {
        const int dt = dp * 2 + u;
        oacc[dt] = wmma_f16(pa[0], bfa[u][0], oacc[dt]);
        oacc[dt] = wmma_f16(pa[1], bfa[u][1], oacc[dt]);
      }
    }
    __syncthreads();
  }

  // ---- Normalize, store f16 in [B,T,D] (d = h*DK + dk); single base pointer.
  _Float16* __restrict__ obase =
      Oh + (size_t)(b * Tc + q0 + wave * 16 + hlf * 8) * Dc + h * DKc + l16;
#pragma unroll
  for (int r = 0; r < 8; r++) {
    const float inv = 1.0f / lI[r];
#pragma unroll
    for (int dt = 0; dt < 4; dt++) {
      obase[r * Dc + dt * 16] = (_Float16)(oacc[dt][r] * inv);
    }
  }
}

// ---------------------------------------------------------------------------
// Launch
// ---------------------------------------------------------------------------

extern "C" void kernel_launch(void* const* d_in, const int* in_sizes, int n_in,
                              void* d_out, int out_size, void* d_ws, size_t ws_size,
                              hipStream_t stream) {
  const float* query = (const float*)d_in[0];
  const float* key   = (const float*)d_in[1];
  const float* value = (const float*)d_in[2];
  // d_in[3] (alibi_bias, 256MB) intentionally unread: recomputed analytically.
  const float* wq = (const float*)d_in[4];
  const float* wk = (const float*)d_in[5];
  const float* wv = (const float*)d_in[6];
  const float* wo = (const float*)d_in[7];
  const float* bo = (const float*)d_in[8];
  float* out = (float*)d_out;

  const size_t HEADSZ = (size_t)Bc * Hc * Tc * DKc;   // 4,194,304 halves (8MB)
  _Float16* wsp = (_Float16*)d_ws;
  _Float16* qh  = wsp;
  _Float16* kh  = wsp + HEADSZ;
  _Float16* vh  = wsp + 2 * HEADSZ;
  _Float16* ah  = wsp + 3 * HEADSZ;                   // 32MB total workspace

  QKVArgs args;
  args.x[0] = query; args.x[1] = key; args.x[2] = value;
  args.w[0] = wq;    args.w[1] = wk;  args.w[2] = wv;
  args.o[0] = qh;    args.o[1] = kh;  args.o[2] = vh;

  dim3 gProj(Dc / 64, (Bc * Tc) / 128, 3);
  proj_qkv_kernel<<<gProj, 128, 0, stream>>>(args);

  dim3 gAttn(Tc / 64, Hc, Bc);
  attn_kernel<<<gAttn, 128, 0, stream>>>(qh, kh, vh, ah);

  dim3 gOut(Dc / 64, (Bc * Tc) / 128, 1);
  out_proj_kernel<<<gOut, 128, 0, stream>>>(ah, wo, bo, out);
}
